// VectorQuantizer_335007449372
// MI455X (gfx1250) — matrix operand order repaired and checked by reading, not compiled, not run on hardware
//
#include <hip/hip_runtime.h>
#include <hip/hip_bf16.h>

typedef __attribute__((ext_vector_type(16))) _Float16 v16h;
typedef __attribute__((ext_vector_type(8)))  _Float16 v8h;
typedef __attribute__((ext_vector_type(8)))  float    v8f;

#define VQ_NE    8192
#define VQ_DIM   256
#define VQ_N     16384            // 16*32*32
#define VQ_SCALE 1024.0f
#define TILE_P   264              // padded row stride in halves (bank swizzle)

// ---------------------------------------------------------------------------
// Kernel 1: codebook prep. One wave per code row:
//   eh[row][c]  = (f16)(e_w[row][c] * S)
//   nse[row]    = -0.5 * S * sum_c e_w[row][c]^2   (folded into WMMA C-init)
//   counts[row] = 0
// ---------------------------------------------------------------------------
__global__ void __launch_bounds__(256) vq_prep_kernel(
    const float* __restrict__ e_w, _Float16* __restrict__ eh,
    float* __restrict__ nse, int* __restrict__ counts)
{
    const int lane = threadIdx.x & 31;
    const int row  = blockIdx.x * (blockDim.x >> 5) + (threadIdx.x >> 5);
    const float*  src = e_w + (size_t)row * VQ_DIM;
    _Float16*     dst = eh  + (size_t)row * VQ_DIM;
    float ssum = 0.0f;
#pragma unroll
    for (int i = 0; i < 8; ++i) {
        float v = src[lane + i * 32];
        ssum += v * v;
        dst[lane + i * 32] = (_Float16)(v * VQ_SCALE);
    }
#pragma unroll
    for (int m = 1; m < 32; m <<= 1) ssum += __shfl_xor(ssum, m, 32);
    if (lane == 0) {
        nse[row]    = -0.5f * VQ_SCALE * ssum;
        counts[row] = 0;
    }
}

// ---------------------------------------------------------------------------
// Kernel 2: WMMA argmax, LDS-staged codebook tiles shared by 8 waves.
// Each wave owns 16 z-rows; the workgroup double-buffers 16-code B tiles
// (8 KB f16) in LDS so the 4 MB codebook is read from L2 once per
// workgroup (128 blocks -> 0.5 GB L2 traffic) instead of once per wave.
// B fragments are software-pipelined one step ahead so each WMMA waits
// only on an older ds_load pair (dscnt<=2), not a full LDS drain.
// acc = z_f16 (16x32) x (S*e)_f16 (32x16) + (-0.5*S*||e||^2)
//     = 0.5*S*(2 z.e - ||e||^2)  -> maximize == argmin distance.
// ---------------------------------------------------------------------------
__global__ void __launch_bounds__(256) vq_argmax_kernel(
    const float* __restrict__ z, const _Float16* __restrict__ eh,
    const float* __restrict__ nse, int* __restrict__ indices,
    int* __restrict__ counts)
{
    __shared__ _Float16 tile[2][16 * TILE_P];   // 2 x 8448 B

    const int lane = threadIdx.x & 31;
    const int wave = threadIdx.x >> 5;              // 0..7
    const int n0   = (blockIdx.x * 8 + wave) * 16;  // first z-row of this wave
    const int b    = n0 >> 10;                      // batch
    const int hw   = (n0 & 1023) + (lane & 15);     // h*32+w for lane's row
    const int kbase = (lane & 16) ? 8 : 0;          // A-matrix K sub-block
    const float* zb = z + ((size_t)b << 18) + hw;

    // staging assignment: each thread copies 16 consecutive halves (32 B)
    const int trow = threadIdx.x >> 4;   // 0..15 code row within tile
    const int tseg = threadIdx.x & 15;   // 16-half segment within row

    // Build A fragments (16x32 f16, ISA layout) directly from NCHW tensor.
    // Lanes 0-15 / 16-31 each share the channel index -> 64B coalesced loads.
    v16h a[8];
#pragma unroll
    for (int f = 0; f < 8; ++f) {
#pragma unroll
        for (int e = 0; e < 16; ++e) {
            int c = 32 * f + kbase + e + ((e >= 8) ? 8 : 0);
            a[f][e] = (_Float16)zb[(size_t)c << 10];
        }
    }

    float best[8];
    int   bidx[8];
#pragma unroll
    for (int r = 0; r < 8; ++r) { best[r] = -3.0e38f; bidx[r] = 0; }

    const int ncol = lane & 15;   // code column within 16-wide tile
    const int koff = (lane & 16) >> 1;   // B-matrix K sub-block per half

    // stage tile 0 into buffer 0
    {
        const int4* gsrc = (const int4*)(eh + (size_t)trow * VQ_DIM + tseg * 16);
        int4* ldst = (int4*)(&tile[0][trow * TILE_P + tseg * 16]);
        ldst[0] = gsrc[0];
        ldst[1] = gsrc[1];
    }
    __syncthreads();

    union U { v16h v; v8h h[2]; };

    for (int t = 0; t < VQ_NE / 16; ++t) {
        const int cur = t & 1;
        // prefetch next tile into the other buffer (overlaps WMMA chain)
        if (t + 1 < VQ_NE / 16) {
            const int4* gsrc = (const int4*)
                (eh + (size_t)((t + 1) * 16 + trow) * VQ_DIM + tseg * 16);
            int4* ldst = (int4*)(&tile[cur ^ 1][trow * TILE_P + tseg * 16]);
            ldst[0] = gsrc[0];
            ldst[1] = gsrc[1];
        }

        const int n = t * 16;
        const float s = nse[n + ncol];          // depends only on N -> C init
        v8f acc;
#pragma unroll
        for (int r = 0; r < 8; ++r) acc[r] = s;

        const _Float16* bp = &tile[cur][ncol * TILE_P + koff];

        // software-pipelined B fragments: load f+1 before WMMA f
        U u[2];
        u[0].h[0] = *(const v8h*)(bp);
        u[0].h[1] = *(const v8h*)(bp + 16);
#pragma unroll
        for (int f = 0; f < 8; ++f) {
            const int pb = f & 1;
            if (f < 7) {
                u[pb ^ 1].h[0] = *(const v8h*)(bp + (f + 1) * 32);
                u[pb ^ 1].h[1] = *(const v8h*)(bp + (f + 1) * 32 + 16);
            }
            acc = __builtin_amdgcn_wmma_f32_16x16x32_f16(
                false, a[f], false, u[pb].v, (short)0, acc, false, false);
        }

        const int code = n + ncol;
#pragma unroll
        for (int r = 0; r < 8; ++r) {
            if (acc[r] > best[r]) { best[r] = acc[r]; bidx[r] = code; }
        }
        __syncthreads();
    }

    // Butterfly argmax across the 16 lanes of each half (rows 0-7 / 8-15).
#pragma unroll
    for (int r = 0; r < 8; ++r) {
        float bv = best[r]; int bi = bidx[r];
#pragma unroll
        for (int m = 1; m <= 8; m <<= 1) {
            float ov = __shfl_xor(bv, m, 32);
            int   oi = __shfl_xor(bi, m, 32);
            if (ov > bv || (ov == bv && oi < bi)) { bv = ov; bi = oi; }
        }
        best[r] = bv; bidx[r] = bi;
    }
    if ((lane & 15) == 0) {
        const int rowbase = n0 + ((lane & 16) ? 8 : 0);
#pragma unroll
        for (int r = 0; r < 8; ++r) {
            indices[rowbase + r] = bidx[r];
            atomicAdd(&counts[bidx[r]], 1);
        }
    }
}

// ---------------------------------------------------------------------------
// Kernel 3: zero the one-hot region (8-byte aligned base -> float2 stores)
// ---------------------------------------------------------------------------
__global__ void vq_zero_kernel(float2* __restrict__ p, long long n2)
{
    long long i      = (long long)blockIdx.x * blockDim.x + threadIdx.x;
    long long stride = (long long)gridDim.x * blockDim.x;
    const float2 zz  = make_float2(0.0f, 0.0f);
    for (; i < n2; i += stride) p[i] = zz;
}

// ---------------------------------------------------------------------------
// Kernel 4: z_q gather (to NCHW output) + per-block squared-diff partials
// ---------------------------------------------------------------------------
__global__ void __launch_bounds__(256) vq_zq_loss_kernel(
    const float* __restrict__ z, const float* __restrict__ e_w,
    const int* __restrict__ indices, float* __restrict__ zq_out,
    float* __restrict__ partials)
{
    __shared__ float sdata[256];
    const int gid = blockIdx.x * 256 + threadIdx.x;  // flat (b,c,h,w)
    const int w = gid & 31, h = (gid >> 5) & 31;
    const int c = (gid >> 10) & 255, b = gid >> 18;
    const int n = (b << 10) + (h << 5) + w;
    const int idx = indices[n];
    const float q  = e_w[(size_t)idx * VQ_DIM + c];
    const float zv = z[gid];
    zq_out[gid] = q;                 // z_q_st == z_q numerically
    const float d = q - zv;
    sdata[threadIdx.x] = d * d;
    __syncthreads();
    for (int s = 128; s > 0; s >>= 1) {
        if (threadIdx.x < s) sdata[threadIdx.x] += sdata[threadIdx.x + s];
        __syncthreads();
    }
    if (threadIdx.x == 0) partials[blockIdx.x] = sdata[0];
}

// ---------------------------------------------------------------------------
// Kernel 5: scatter one-hot + float indices output
// ---------------------------------------------------------------------------
__global__ void vq_scatter_kernel(const int* __restrict__ indices,
                                  float* __restrict__ onehot,
                                  float* __restrict__ idx_out)
{
    const int n = blockIdx.x * blockDim.x + threadIdx.x;
    if (n < VQ_N) {
        const int idx = indices[n];
        onehot[(size_t)n * VQ_NE + idx] = 1.0f;
        idx_out[n] = (float)idx;
    }
}

// ---------------------------------------------------------------------------
// Kernel 6: deterministic finalize: loss + perplexity
// ---------------------------------------------------------------------------
__global__ void __launch_bounds__(256) vq_finalize_kernel(
    const float* __restrict__ partials, int nPart,
    const int* __restrict__ counts,
    float* __restrict__ loss_out, float* __restrict__ perp_out)
{
    __shared__ float sdata[256];
    float s = 0.0f;
    for (int i = threadIdx.x; i < nPart; i += 256) s += partials[i];
    sdata[threadIdx.x] = s;
    __syncthreads();
    for (int st = 128; st > 0; st >>= 1) {
        if (threadIdx.x < st) sdata[threadIdx.x] += sdata[threadIdx.x + st];
        __syncthreads();
    }
    if (threadIdx.x == 0)
        loss_out[0] = 1.25f * sdata[0] / 4194304.0f;  // (1+beta)*mean
    __syncthreads();
    float hsum = 0.0f;
    for (int k = threadIdx.x; k < VQ_NE; k += 256) {
        float p = (float)counts[k] / (float)VQ_N;
        hsum += p * logf(p + 1e-10f);
    }
    sdata[threadIdx.x] = hsum;
    __syncthreads();
    for (int st = 128; st > 0; st >>= 1) {
        if (threadIdx.x < st) sdata[threadIdx.x] += sdata[threadIdx.x + st];
        __syncthreads();
    }
    if (threadIdx.x == 0) perp_out[0] = expf(-sdata[0]);
}

// ---------------------------------------------------------------------------
extern "C" void kernel_launch(void* const* d_in, const int* in_sizes, int n_in,
                              void* d_out, int out_size, void* d_ws, size_t ws_size,
                              hipStream_t stream)
{
    (void)in_sizes; (void)n_in; (void)out_size; (void)ws_size;
    const float* z   = (const float*)d_in[0];   // (16,256,32,32) f32
    const float* e_w = (const float*)d_in[1];   // (8192,256) f32
    float* out = (float*)d_out;

    // workspace carve (~4.4 MB)
    char* ws = (char*)d_ws;
    _Float16* eh      = (_Float16*)ws;                                  // 4 MB
    float*    nse     = (float*)(ws + 4194304);                         // 32 KB
    int*      indices = (int*)  (ws + 4194304 + 32768);                 // 64 KB
    int*      counts  = (int*)  (ws + 4194304 + 32768 + 65536);         // 32 KB
    float*    partials= (float*)(ws + 4194304 + 32768 + 65536 + 32768); // 64 KB

    // output layout: loss | z_q_st (4194304) | perplexity | one_hot (16384*8192) | indices (16384)
    float* loss_out = out;
    float* zq_out   = out + 1;
    float* perp_out = out + 1 + 4194304;
    float* onehot   = out + 2 + 4194304;
    float* idx_out  = out + 2 + 4194304 + (long long)VQ_N * VQ_NE;

    vq_prep_kernel   <<<1024, 256, 0, stream>>>(e_w, eh, nse, counts);
    vq_argmax_kernel <<<128, 256, 0, stream>>>(z, eh, nse, indices, counts);
    vq_zero_kernel   <<<32768, 256, 0, stream>>>((float2*)onehot,
                                                 ((long long)VQ_N * VQ_NE) / 2);
    vq_zq_loss_kernel<<<16384, 256, 0, stream>>>(z, e_w, indices, zq_out, partials);
    vq_scatter_kernel<<<64, 256, 0, stream>>>(indices, onehot, idx_out);
    vq_finalize_kernel<<<1, 256, 0, stream>>>(partials, 16384, counts,
                                              loss_out, perp_out);
}
